// CausalSelfAttention_5360119185605
// MI455X (gfx1250) — compile-verified, matrix operand order checked by reading, not run on hardware
//
#include <hip/hip_runtime.h>

#define TSEQ 2048
#define CDIM 1024
#define NHEAD 16
#define HSD 64
#define BSZ 4

typedef __attribute__((ext_vector_type(16))) __bf16 v16bf;
typedef __attribute__((ext_vector_type(8)))  __bf16 v8bf;
typedef __attribute__((ext_vector_type(8)))  float  v8f;

static __device__ __forceinline__ __bf16 f2bf(float f) {
    unsigned u = __builtin_bit_cast(unsigned, f);
    u += 0x7FFFu + ((u >> 16) & 1u);               // round-to-nearest-even
    unsigned short s = (unsigned short)(u >> 16);
    return __builtin_bit_cast(__bf16, s);
}

union ABFrag { v16bf v; v8bf h[2]; };

#define WMMA_BF16(A, B, C) \
    __builtin_amdgcn_wmma_f32_16x16x32_bf16(false, (A), false, (B), (short)0, (C), false, false)

// ---------------------------------------------------------------- conversions
__global__ __launch_bounds__(256) void cvt_f32_to_bf16(const float* __restrict__ in,
                                                       __bf16* __restrict__ out, int n) {
    for (int i = blockIdx.x * blockDim.x + threadIdx.x; i < n; i += gridDim.x * blockDim.x)
        out[i] = f2bf(in[i]);
}

// out[n*rows + k] = in[k*cols + n]   (weights -> bf16, transposed to N-major)
__global__ __launch_bounds__(256) void cvt_transpose_bf16(const float* __restrict__ in,
                                                          __bf16* __restrict__ out,
                                                          int rows, int cols) {
    int total = rows * cols;
    for (int i = blockIdx.x * blockDim.x + threadIdx.x; i < total; i += gridDim.x * blockDim.x) {
        int n = i / rows;
        int k = i - n * rows;
        out[i] = f2bf(in[k * cols + n]);
    }
}

// ---------------------------------------------------------------- GEMM core
// Per-wave tile: 32 (M) x 64 (N).  2 A-frags x 4 B-frags -> 8 WMMAs per k-step.
struct Frags { ABFrag a[2]; ABFrag b[4]; };

static __device__ __forceinline__ void load_frags(Frags& f,
                                                  const __bf16* __restrict__ arow0,
                                                  const __bf16* __restrict__ arow1,
                                                  const __bf16* __restrict__ brow0,
                                                  int k0, int kh) {
    f.a[0].h[0] = *(const v8bf*)(arow0 + k0 + kh * 8);
    f.a[0].h[1] = *(const v8bf*)(arow0 + k0 + 16 + kh * 8);
    f.a[1].h[0] = *(const v8bf*)(arow1 + k0 + kh * 8);
    f.a[1].h[1] = *(const v8bf*)(arow1 + k0 + 16 + kh * 8);
#pragma unroll
    for (int j = 0; j < 4; ++j) {
        const __bf16* br = brow0 + (size_t)(j * 16) * CDIM;
        f.b[j].h[0] = *(const v8bf*)(br + k0 + kh * 8);
        f.b[j].h[1] = *(const v8bf*)(br + k0 + 16 + kh * 8);
    }
}

static __device__ __forceinline__ void mma_frags(v8f acc[2][4], const Frags& f) {
#pragma unroll
    for (int mi = 0; mi < 2; ++mi)
#pragma unroll
        for (int j = 0; j < 4; ++j)
            acc[mi][j] = WMMA_BF16(f.a[mi].v, f.b[j].v, acc[mi][j]);
}

// ---------------------------------------------------------------- QKV GEMM
// X: [8192][1024] bf16. Wt: [3072][1024] bf16 (N-major over K).
// Writes Q,K as [B*NH][T][HS] bf16 and V transposed [B*NH][HS][T] bf16, bias fused.
__global__ __launch_bounds__(256) void gemm_qkv(const __bf16* __restrict__ X,
                                                const __bf16* __restrict__ Wt,
                                                const float* __restrict__ bias,
                                                __bf16* __restrict__ Q,
                                                __bf16* __restrict__ Kc,
                                                __bf16* __restrict__ Vt) {
    const int lane = threadIdx.x & 31;
    const int wave = threadIdx.x >> 5;
    const int col  = lane & 15;
    const int kh   = lane >> 4;
    const int m0 = blockIdx.y * 128 + (wave >> 1) * 32;    // 4 waves in M
    const int n0 = blockIdx.x * 128 + (wave & 1) * 64;     // 2 waves in N
    const v8f vz = {0.f, 0.f, 0.f, 0.f, 0.f, 0.f, 0.f, 0.f};
    v8f acc[2][4] = {{vz, vz, vz, vz}, {vz, vz, vz, vz}};

    const __bf16* arow0 = X  + (size_t)(m0 + col) * CDIM;
    const __bf16* arow1 = arow0 + (size_t)16 * CDIM;
    const __bf16* brow0 = Wt + (size_t)(n0 + col) * CDIM;

    Frags f0, f1;
    load_frags(f0, arow0, arow1, brow0, 0, kh);
    for (int k0 = 0; k0 < CDIM; k0 += 64) {                // ping-pong pipeline
        load_frags(f1, arow0, arow1, brow0, k0 + 32, kh);
        __builtin_prefetch(arow0 + k0 + 256, 0, 0);
        __builtin_prefetch(brow0 + k0 + 256, 0, 0);
        mma_frags(acc, f0);
        if (k0 + 64 < CDIM)
            load_frags(f0, arow0, arow1, brow0, k0 + 64, kh);
        mma_frags(acc, f1);
    }

#pragma unroll
    for (int j = 0; j < 4; ++j) {
        int c  = n0 + j * 16 + col;
        float bv = bias[c];
        int seg = c >> 10;             // 0=Q 1=K 2=V
        int cc  = c & (CDIM - 1);
        int h   = cc >> 6;
        int d   = cc & (HSD - 1);
#pragma unroll
        for (int mi = 0; mi < 2; ++mi)
#pragma unroll
            for (int i = 0; i < 8; ++i) {
                int m  = m0 + mi * 16 + i + 8 * kh;
                int bb = m >> 11;
                int t  = m & (TSEQ - 1);
                int bh = (bb << 4) | h;
                __bf16 v = f2bf(acc[mi][j][i] + bv);
                if (seg == 0)      Q [((size_t)bh * TSEQ + t) * HSD + d] = v;
                else if (seg == 1) Kc[((size_t)bh * TSEQ + t) * HSD + d] = v;
                else               Vt[((size_t)bh * HSD + d) * TSEQ + t] = v;
            }
    }
}

// ---------------------------------------------------------------- flash attention
// One wave per (b,h,q-tile of 16). Online softmax over 32-wide k-chunks.
__global__ __launch_bounds__(32) void attn_fwd(const __bf16* __restrict__ Q,
                                               const __bf16* __restrict__ Kc,
                                               const __bf16* __restrict__ Vt,
                                               __bf16* __restrict__ Y) {
    const int lane = threadIdx.x;
    const int col  = lane & 15;
    const int kh   = lane >> 4;
    const int qt   = blockIdx.x & (TSEQ / 16 - 1);   // 128 q-tiles
    const int bh   = blockIdx.x >> 7;                // 0..63
    const int bidx = bh >> 4;
    const int h    = bh & 15;
    const int q0   = qt * 16;

    __shared__ __align__(16) float  sS[16][32];
    __shared__ __align__(16) __bf16 sP[16][32];
    __shared__ float sAlpha[16];
    __shared__ float sNorm[16];

    // Q fragments (16 rows x 64 d), resident for whole block
    const __bf16* qrow = Q + ((size_t)bh * TSEQ + q0 + col) * HSD;
    ABFrag aq0, aq1;
    aq0.h[0] = *(const v8bf*)(qrow + kh * 8);
    aq0.h[1] = *(const v8bf*)(qrow + 16 + kh * 8);
    aq1.h[0] = *(const v8bf*)(qrow + 32 + kh * 8);
    aq1.h[1] = *(const v8bf*)(qrow + 48 + kh * 8);

    const v8f vz = {0.f, 0.f, 0.f, 0.f, 0.f, 0.f, 0.f, 0.f};
    v8f o0 = vz, o1 = vz, o2 = vz, o3 = vz;        // O: 16 q x 64 d fp32
    float rowM = -1e30f, rowL = 0.f;               // per-row state (lanes 0..15)

    const int nchunks = q0 / 32 + 1;               // causal: skip fully-masked chunks
    for (int ch = 0; ch < nchunks; ++ch) {
        const int kt = ch * 32;

        // ---- hoist V fragment loads: in flight during score WMMAs + softmax
        const __bf16* vbase = Vt + ((size_t)bh * HSD + col) * TSEQ + kt;
        ABFrag bv0, bv1, bv2, bv3;
        bv0.h[0] = *(const v8bf*)(vbase + kh * 8);
        bv0.h[1] = *(const v8bf*)(vbase + 16 + kh * 8);
        bv1.h[0] = *(const v8bf*)(vbase + (size_t)16 * TSEQ + kh * 8);
        bv1.h[1] = *(const v8bf*)(vbase + (size_t)16 * TSEQ + 16 + kh * 8);
        bv2.h[0] = *(const v8bf*)(vbase + (size_t)32 * TSEQ + kh * 8);
        bv2.h[1] = *(const v8bf*)(vbase + (size_t)32 * TSEQ + 16 + kh * 8);
        bv3.h[0] = *(const v8bf*)(vbase + (size_t)48 * TSEQ + kh * 8);
        bv3.h[1] = *(const v8bf*)(vbase + (size_t)48 * TSEQ + 16 + kh * 8);

        // ---- scores S = Q @ K^T for 32 k-columns (two 16-col tiles)
        const __bf16* krow0 = Kc + ((size_t)bh * TSEQ + kt + col) * HSD;
        const __bf16* krow1 = krow0 + (size_t)16 * HSD;
        ABFrag bka0, bka1, bkb0, bkb1;
        bka0.h[0] = *(const v8bf*)(krow0 + kh * 8);
        bka0.h[1] = *(const v8bf*)(krow0 + 16 + kh * 8);
        bka1.h[0] = *(const v8bf*)(krow0 + 32 + kh * 8);
        bka1.h[1] = *(const v8bf*)(krow0 + 48 + kh * 8);
        bkb0.h[0] = *(const v8bf*)(krow1 + kh * 8);
        bkb0.h[1] = *(const v8bf*)(krow1 + 16 + kh * 8);
        bkb1.h[0] = *(const v8bf*)(krow1 + 32 + kh * 8);
        bkb1.h[1] = *(const v8bf*)(krow1 + 48 + kh * 8);
        v8f s0 = vz, s1 = vz;
        s0 = WMMA_BF16(aq0.v, bka0.v, s0);
        s0 = WMMA_BF16(aq1.v, bka1.v, s0);
        s1 = WMMA_BF16(aq0.v, bkb0.v, s1);
        s1 = WMMA_BF16(aq1.v, bkb1.v, s1);

        // ---- scale + causal mask, spill to LDS (D-layout: lane=col, vgpr=row)
        const int kc0 = kt + col;
        const int kc1 = kt + 16 + col;
#pragma unroll
        for (int i = 0; i < 8; ++i) {
            int qg = q0 + i + 8 * kh;
            float v0 = s0[i] * 0.125f;             // 1/sqrt(64)
            float v1 = s1[i] * 0.125f;
            if (kc0 > qg) v0 = -1e30f;
            if (kc1 > qg) v1 = -1e30f;
            sS[i + 8 * kh][col]      = v0;
            sS[i + 8 * kh][16 + col] = v1;
        }
        __syncthreads();

        // ---- online softmax, one row per lane (lanes 0..15)
        if (lane < 16) {
            float cm = -1e30f;
#pragma unroll
            for (int j = 0; j < 32; ++j) cm = fmaxf(cm, sS[lane][j]);
            float nM    = fmaxf(rowM, cm);
            float alpha = __expf(rowM - nM);
            float sum   = 0.f;
#pragma unroll
            for (int j = 0; j < 32; ++j) {
                float p = __expf(sS[lane][j] - nM);
                sum += p;
                sP[lane][j] = f2bf(p);
            }
            rowL = rowL * alpha + sum;
            rowM = nM;
            sAlpha[lane] = alpha;
        }
        __syncthreads();

        // ---- rescale O, then O += P @ V_chunk (4 WMMAs over 64 d)
        {
            float al0 = sAlpha[0 + 8 * kh], al1 = sAlpha[1 + 8 * kh];
            float al2 = sAlpha[2 + 8 * kh], al3 = sAlpha[3 + 8 * kh];
            float al4 = sAlpha[4 + 8 * kh], al5 = sAlpha[5 + 8 * kh];
            float al6 = sAlpha[6 + 8 * kh], al7 = sAlpha[7 + 8 * kh];
            o0[0]*=al0; o0[1]*=al1; o0[2]*=al2; o0[3]*=al3; o0[4]*=al4; o0[5]*=al5; o0[6]*=al6; o0[7]*=al7;
            o1[0]*=al0; o1[1]*=al1; o1[2]*=al2; o1[3]*=al3; o1[4]*=al4; o1[5]*=al5; o1[6]*=al6; o1[7]*=al7;
            o2[0]*=al0; o2[1]*=al1; o2[2]*=al2; o2[3]*=al3; o2[4]*=al4; o2[5]*=al5; o2[6]*=al6; o2[7]*=al7;
            o3[0]*=al0; o3[1]*=al1; o3[2]*=al2; o3[3]*=al3; o3[4]*=al4; o3[5]*=al5; o3[6]*=al6; o3[7]*=al7;
        }
        ABFrag ap;                                  // A-layout read of P from LDS
        ap.h[0] = *(const v8bf*)&sP[col][kh * 8];
        ap.h[1] = *(const v8bf*)&sP[col][16 + kh * 8];

        o0 = WMMA_BF16(ap.v, bv0.v, o0);
        o1 = WMMA_BF16(ap.v, bv1.v, o1);
        o2 = WMMA_BF16(ap.v, bv2.v, o2);
        o3 = WMMA_BF16(ap.v, bv3.v, o3);
        __syncthreads();
    }

    if (lane < 16) sNorm[lane] = 1.0f / rowL;
    __syncthreads();

    // Y[b][q][h*64 + d] bf16 row-major (C-contiguous) for the output projection
#pragma unroll
    for (int i = 0; i < 8; ++i) {
        int qg = q0 + i + 8 * kh;
        float nrm = sNorm[i + 8 * kh];
        size_t base = ((size_t)(bidx * TSEQ + qg)) * CDIM + h * HSD + col;
        Y[base +  0] = f2bf(o0[i] * nrm);
        Y[base + 16] = f2bf(o1[i] * nrm);
        Y[base + 32] = f2bf(o2[i] * nrm);
        Y[base + 48] = f2bf(o3[i] * nrm);
    }
}

// ---------------------------------------------------------------- output projection
// Yb: [8192][1024] bf16, Wt: [1024][1024] bf16 (N-major), out fp32 + bias
__global__ __launch_bounds__(256) void gemm_proj(const __bf16* __restrict__ Yb,
                                                 const __bf16* __restrict__ Wt,
                                                 const float* __restrict__ bias,
                                                 float* __restrict__ out) {
    const int lane = threadIdx.x & 31;
    const int wave = threadIdx.x >> 5;
    const int col  = lane & 15;
    const int kh   = lane >> 4;
    const int m0 = blockIdx.y * 128 + (wave >> 1) * 32;
    const int n0 = blockIdx.x * 128 + (wave & 1) * 64;
    const v8f vz = {0.f, 0.f, 0.f, 0.f, 0.f, 0.f, 0.f, 0.f};
    v8f acc[2][4] = {{vz, vz, vz, vz}, {vz, vz, vz, vz}};

    const __bf16* arow0 = Yb + (size_t)(m0 + col) * CDIM;
    const __bf16* arow1 = arow0 + (size_t)16 * CDIM;
    const __bf16* brow0 = Wt + (size_t)(n0 + col) * CDIM;

    Frags f0, f1;
    load_frags(f0, arow0, arow1, brow0, 0, kh);
    for (int k0 = 0; k0 < CDIM; k0 += 64) {
        load_frags(f1, arow0, arow1, brow0, k0 + 32, kh);
        __builtin_prefetch(arow0 + k0 + 256, 0, 0);
        __builtin_prefetch(brow0 + k0 + 256, 0, 0);
        mma_frags(acc, f0);
        if (k0 + 64 < CDIM)
            load_frags(f0, arow0, arow1, brow0, k0 + 64, kh);
        mma_frags(acc, f1);
    }

#pragma unroll
    for (int j = 0; j < 4; ++j) {
        int c = n0 + j * 16 + col;
        float bv = bias[c];
#pragma unroll
        for (int mi = 0; mi < 2; ++mi)
#pragma unroll
            for (int i = 0; i < 8; ++i) {
                int m = m0 + mi * 16 + i + 8 * kh;
                out[(size_t)m * CDIM + c] = acc[mi][j][i] + bv;
            }
    }
}

// ---------------------------------------------------------------- launch
extern "C" void kernel_launch(void* const* d_in, const int* in_sizes, int n_in,
                              void* d_out, int out_size, void* d_ws, size_t ws_size,
                              hipStream_t stream) {
    const float* x      = (const float*)d_in[0];
    const float* w_attn = (const float*)d_in[1];
    const float* b_attn = (const float*)d_in[2];
    const float* w_proj = (const float*)d_in[3];
    const float* b_proj = (const float*)d_in[4];
    float* out = (float*)d_out;

    char* ws = (char*)d_ws;
    size_t off = 0;
    auto carve = [&](size_t bytes) -> char* {
        char* p = ws + off;
        off += (bytes + 255) & ~(size_t)255;
        return p;
    };
    const size_t MT = (size_t)BSZ * TSEQ;                 // 8192 rows
    __bf16* Xbf    = (__bf16*)carve(MT * CDIM * 2);
    __bf16* WattnT = (__bf16*)carve((size_t)3 * CDIM * CDIM * 2);
    __bf16* WprojT = (__bf16*)carve((size_t)CDIM * CDIM * 2);
    __bf16* Qbf    = (__bf16*)carve(MT * CDIM * 2);
    __bf16* Kbf    = (__bf16*)carve(MT * CDIM * 2);
    __bf16* Vtb    = (__bf16*)carve(MT * CDIM * 2);
    __bf16* Ybf    = (__bf16*)carve(MT * CDIM * 2);

    cvt_f32_to_bf16<<<8192, 256, 0, stream>>>(x, Xbf, (int)(MT * CDIM));
    cvt_transpose_bf16<<<4096, 256, 0, stream>>>(w_attn, WattnT, CDIM, 3 * CDIM);
    cvt_transpose_bf16<<<2048, 256, 0, stream>>>(w_proj, WprojT, CDIM, CDIM);

    gemm_qkv<<<dim3(3 * CDIM / 128, MT / 128), 256, 0, stream>>>(
        Xbf, WattnT, b_attn, Qbf, Kbf, Vtb);

    attn_fwd<<<BSZ * NHEAD * (TSEQ / 16), 32, 0, stream>>>(Qbf, Kbf, Vtb, Ybf);

    gemm_proj<<<dim3(CDIM / 128, MT / 128), 256, 0, stream>>>(
        Ybf, WprojT, b_proj, out);
}
